// FinalLayer_14448269984096
// MI455X (gfx1250) — matrix operand order repaired and checked by reading, not compiled, not run on hardware
//
#include <hip/hip_runtime.h>
#include <hip/hip_bf16.h>
#include <math.h>

typedef __attribute__((ext_vector_type(16))) __bf16 v16bf;
typedef __attribute__((ext_vector_type(8)))  __bf16 v8bf;
typedef __attribute__((ext_vector_type(8)))  float  v8f;

#define HIDDEN   2048
#define NB       8
#define HROWS    4096
#define MROWS    (NB * HROWS)   // 32768
#define NOUT     256
#define PROJ_IN  1024
#define BM       128
#define BN       64
#define BK       32
#define LDX      36             // x stage stride (floats), 144B rows (16B aligned)
#define LDA      40             // A stride (ushorts), 80B rows (16B aligned)
#define LDB      40             // B stride (ushorts)

__device__ __forceinline__ unsigned short bf16_rn(float f) {
  unsigned int u = __float_as_uint(f);
  u += 0x7FFFu + ((u >> 16) & 1u);           // round-to-nearest-even
  return (unsigned short)(u >> 16);
}
__device__ __forceinline__ float bf16_f(unsigned short h) {
  return __uint_as_float(((unsigned int)h) << 16);
}
// Build a v16bf A/B fragment from two 16-byte LDS chunks at p and p+gap.
__device__ __forceinline__ v16bf ld16gap(const unsigned short* p, int gap) {
  v8bf a = *(const v8bf*)p;
  v8bf b = *(const v8bf*)(p + gap);
  v16bf r;
#pragma unroll
  for (int i = 0; i < 8; ++i) { r[i] = a[i]; r[i + 8] = b[i]; }
  return r;
}

// ---------------------------------------------------------------------------
// Kernel 1: ftime = silu(c) @ W_proj.T + b_proj, then build per-batch 16x16
// circulant matrices C_b[n,m] = (1/16) sum_v ftime[b,v] * cos(2*pi*v*((n-m)&15)/16)
// ---------------------------------------------------------------------------
__global__ void __launch_bounds__(128) adafm_prep(const float* __restrict__ c,
                                                  const float* __restrict__ Wp,
                                                  const float* __restrict__ bp,
                                                  float* __restrict__ Cmat) {
  __shared__ float ft[NB][16];
  const int t = threadIdx.x;              // 128 threads: (b, v)
  const int b = t >> 4, v = t & 15;
  float acc = bp[v];
  for (int j = 0; j < PROJ_IN; ++j) {
    float cv = c[b * PROJ_IN + j];
    acc += (cv / (1.f + __expf(-cv))) * Wp[v * PROJ_IN + j];
  }
  ft[b][v] = acc;
  __syncthreads();
  for (int idx = t; idx < NB * 256; idx += 128) {
    const int bb = idx >> 8, n = (idx >> 4) & 15, m = idx & 15;
    const int d = (n - m) & 15;
    float g = 0.f;
#pragma unroll
    for (int vv = 0; vv < 16; ++vv)
      g += ft[bb][vv] * __cosf(0.39269908169872414f * (float)(vv * d)); // 2*pi/16
    Cmat[idx] = g * 0.0625f;
  }
}

// ---------------------------------------------------------------------------
// Kernel 2: fused patch-circulant modulation + GEMM (32768x2048 @ 2048x256)
// bf16x2 split (hi+lo) with f32 WMMA accumulation, scaled epilogue.
// ---------------------------------------------------------------------------
__global__ void __launch_bounds__(256) adafm_gemm(const float* __restrict__ x,
                                                  const float* __restrict__ Wl,
                                                  const float* __restrict__ blin,
                                                  const float* __restrict__ sz,
                                                  const float* __restrict__ Cmat,
                                                  float* __restrict__ out) {
  __shared__ __align__(16) float          xs[BM * LDX];   // 18432 B raw fp32 x tile
  __shared__ __align__(16) unsigned short Ah[BM * LDA];   // 10240 B modulated A hi
  __shared__ __align__(16) unsigned short Al[BM * LDA];   // 10240 B modulated A lo
  __shared__ __align__(16) unsigned short Bh[BN * LDB];   //  5120 B W hi
  __shared__ __align__(16) unsigned short Bl[BN * LDB];   //  5120 B W lo
  __shared__ float Cs[256];                               // circulant for this batch

  const int t       = threadIdx.x;
  const int o0      = blockIdx.x * BN;          // N tile
  const int rowBase = blockIdx.y * BM;          // M tile (128 | 4096 => single batch)
  const int batch   = rowBase >> 12;

  Cs[t] = Cmat[batch * 256 + t];

  const int  lane   = t & 31;
  const int  wid    = t >> 5;                   // 8 waves: 4 in M x 2 in N
  const int  wm     = (wid & 3) * 32;
  const int  wn     = (wid >> 2) * 32;
  const int  l16    = lane & 15;
  const bool hiHalf = lane >= 16;
  const int  aoff   = hiHalf ? 8 : 0;           // A: K-half select per ISA layout
  const int  boff   = hiHalf ? 8 : 0;           // B: K-half select per ISA layout

  v8f acc[2][2];
#pragma unroll
  for (int mi = 0; mi < 2; ++mi)
#pragma unroll
    for (int ni = 0; ni < 2; ++ni)
#pragma unroll
      for (int j = 0; j < 8; ++j) acc[mi][ni][j] = 0.f;

  const int xr = t & 127, xh = t >> 7;          // x stage: row, col-half
  const int bn = t & 63,  bg = t >> 6;          // W stage: out-row, col-group
  const float* gx_base = x  + (size_t)(rowBase + xr) * HIDDEN + xh * 16;
  const float* gw_base = Wl + (size_t)(o0 + bn)      * HIDDEN + bg * 8;

  for (int k0 = 0; k0 < HIDDEN; k0 += BK) {
    __syncthreads();  // previous iteration's WMMA done with LDS

    // ---- stage raw x tile (128 x 32 fp32) ----
    {
      const float4* gp = (const float4*)(gx_base + k0);
      float* dst = &xs[xr * LDX + xh * 16];
      float4 a0 = gp[0], a1 = gp[1], a2 = gp[2], a3 = gp[3];
      *(float4*)(dst + 0)  = a0;
      *(float4*)(dst + 4)  = a1;
      *(float4*)(dst + 8)  = a2;
      *(float4*)(dst + 12) = a3;
    }
    // ---- stage W tile (64 x 32) as bf16 hi/lo ----
    {
      const float4* gp = (const float4*)(gw_base + k0);
      float4 w0 = gp[0], w1 = gp[1];
      float wv[8] = {w0.x, w0.y, w0.z, w0.w, w1.x, w1.y, w1.z, w1.w};
      unsigned short* ph = &Bh[bn * LDB + bg * 8];
      unsigned short* pl = &Bl[bn * LDB + bg * 8];
#pragma unroll
      for (int i = 0; i < 8; ++i) {
        unsigned short h = bf16_rn(wv[i]);
        ph[i] = h;
        pl[i] = bf16_rn(wv[i] - bf16_f(h));
      }
    }
    if (k0 + BK < HIDDEN) __builtin_prefetch(gx_base + k0 + BK, 0, 1);
    __syncthreads();

    // ---- modulation: one thread = one 4x4 patch, xm_patch = C_b * x_patch ----
    {
      const int pr = t >> 3, pc = t & 7;        // 32 patch-rows x 8 patch-cols
      float xv[16];
#pragma unroll
      for (int pi = 0; pi < 4; ++pi) {
        float4 rv = *(const float4*)&xs[(pr * 4 + pi) * LDX + pc * 4];
        xv[pi * 4 + 0] = rv.x; xv[pi * 4 + 1] = rv.y;
        xv[pi * 4 + 2] = rv.z; xv[pi * 4 + 3] = rv.w;
      }
#pragma unroll
      for (int n = 0; n < 16; ++n) {
        float y = 0.f;
#pragma unroll
        for (int m = 0; m < 16; ++m) y += Cs[n * 16 + m] * xv[m];
        unsigned short h = bf16_rn(y);
        const int ar = (pr * 4 + (n >> 2)) * LDA + pc * 4 + (n & 3);
        Ah[ar] = h;
        Al[ar] = bf16_rn(y - bf16_f(h));
      }
    }
    __syncthreads();

    // ---- WMMA: acc += Ahi*Bhi + Ahi*Blo + Alo*Bhi ----
    {
      v16bf ah[2], al2[2], bh2[2], bl2[2];
#pragma unroll
      for (int mi = 0; mi < 2; ++mi) {
        const int base = (wm + mi * 16 + l16) * LDA + aoff;
        ah[mi]  = ld16gap(&Ah[base], 16);       // K 0..7 / 16..23 (or +8 half)
        al2[mi] = ld16gap(&Al[base], 16);
      }
#pragma unroll
      for (int ni = 0; ni < 2; ++ni) {
        const int base = (wn + ni * 16 + l16) * LDB + boff;
        bh2[ni] = ld16gap(&Bh[base], 16);       // K 0..7 / 16..23 (or +8 half)
        bl2[ni] = ld16gap(&Bl[base], 16);
      }
#pragma unroll
      for (int mi = 0; mi < 2; ++mi)
#pragma unroll
        for (int ni = 0; ni < 2; ++ni) {
          acc[mi][ni] = __builtin_amdgcn_wmma_f32_16x16x32_bf16(
              false, ah[mi], false, bh2[ni], (short)0, acc[mi][ni], false, false);
          acc[mi][ni] = __builtin_amdgcn_wmma_f32_16x16x32_bf16(
              false, ah[mi], false, bl2[ni], (short)0, acc[mi][ni], false, false);
          acc[mi][ni] = __builtin_amdgcn_wmma_f32_16x16x32_bf16(
              false, al2[mi], false, bh2[ni], (short)0, acc[mi][ni], false, false);
        }
    }
  }

  // ---- epilogue: out = (sz*sqrt(2048)) * (acc + b_lin), C layout per ISA ----
  const float rsc = 45.254833995939045f;  // sqrt(2048)
#pragma unroll
  for (int ni = 0; ni < 2; ++ni) {
    const int O   = o0 + wn + ni * 16 + l16;
    const float s = sz[O] * rsc;
    const float bb = blin[O];
#pragma unroll
    for (int mi = 0; mi < 2; ++mi)
#pragma unroll
      for (int j = 0; j < 8; ++j) {
        const int R = rowBase + wm + mi * 16 + j + (hiHalf ? 8 : 0);
        out[(size_t)R * NOUT + O] = s * (acc[mi][ni][j] + bb);
      }
  }
}

extern "C" void kernel_launch(void* const* d_in, const int* in_sizes, int n_in,
                              void* d_out, int out_size, void* d_ws, size_t ws_size,
                              hipStream_t stream) {
  const float* x    = (const float*)d_in[0];
  const float* c    = (const float*)d_in[1];
  const float* Wp   = (const float*)d_in[2];
  const float* bp   = (const float*)d_in[3];
  const float* Wl   = (const float*)d_in[4];
  const float* blin = (const float*)d_in[5];
  const float* szv  = (const float*)d_in[6];
  float* out  = (float*)d_out;
  float* Cmat = (float*)d_ws;   // 8 * 256 floats = 8 KB

  adafm_prep<<<1, 128, 0, stream>>>(c, Wp, bp, Cmat);
  adafm_gemm<<<dim3(NOUT / BN, MROWS / BM), 256, 0, stream>>>(x, Wl, blin, szv,
                                                              Cmat, out);
  (void)in_sizes; (void)n_in; (void)out_size; (void)ws_size;
}
